// GeneralDeformRoIPoolPack_13469017440353
// MI455X (gfx1250) — matrix operand order repaired and checked by reading, not compiled, not run on hardware
//
#include <hip/hip_runtime.h>
#include <hip/hip_bf16.h>
#include <math.h>

// ---------------- problem constants (fixed by the reference) ----------------
#define BB      2
#define CC      256
#define HH      200
#define WWD     200
#define HWSZ    40000        // HH*WWD
#define ROIS_N  512
#define OUTD    7
#define SP      49           // OUTD*OUTD
#define FC      1024
#define FLATK   12544        // SP*CC

typedef __attribute__((ext_vector_type(16))) __bf16 bf16x16;
typedef __attribute__((ext_vector_type(8)))  float  f32x8;
typedef __attribute__((ext_vector_type(4)))  unsigned int u32x4;
typedef __attribute__((ext_vector_type(8)))  int i32x8;
typedef __attribute__((ext_vector_type(4)))  int i32x4;

// ---------------------------------------------------------------------------
// feat NCHW -> NHWC, LDS-tiled 32x32 transpose (coalesced both sides)
// grid (HWSZ/32, CC/32, BB), block 256
// ---------------------------------------------------------------------------
__global__ __launch_bounds__(256) void feat_transpose(const float* __restrict__ src,
                                                      float* __restrict__ dst) {
  __shared__ float tile[32][33];
  const int b  = blockIdx.z;
  const int c0 = blockIdx.y * 32;
  const int h0 = blockIdx.x * 32;
  const int tx = threadIdx.x & 31;
  const int ty = threadIdx.x >> 5;              // 0..7
  for (int i = 0; i < 4; ++i) {
    int c = c0 + ty + i * 8;
    tile[ty + i * 8][tx] = src[((size_t)b * CC + c) * HWSZ + h0 + tx];
  }
  __syncthreads();
  for (int i = 0; i < 4; ++i) {
    int h = h0 + ty + i * 8;
    dst[((size_t)b * HWSZ + h) * CC + c0 + tx] = tile[tx][ty + i * 8];
  }
}

// ---------------------------------------------------------------------------
// Generic deformable RoI pool. feat_t is NHWC. out is [N, 49, C] (s-major).
// off : nullable, row stride offStride, layout ch*49+s (ch0=w, ch1=h) (raw MLP out)
// tcv : nullable, row stride tcStride; ch0/1 -> rescale via tanh, ch2/3 add to offset
// grid = N*49 blocks, 256 threads (one per channel)
// ---------------------------------------------------------------------------
__global__ __launch_bounds__(256) void deform_pool(
    const float* __restrict__ feat_t, const float* __restrict__ rois,
    const float* __restrict__ off, int offStride,
    const float* __restrict__ tcv, int tcStride,
    float* __restrict__ out) {
  const int n  = blockIdx.x / SP;
  const int s  = blockIdx.x % SP;
  const int ph = s / OUTD, pw = s % OUTD;
  const int c  = threadIdx.x;

  const float* r = rois + (size_t)n * 5;
  const int   b  = (int)r[0];
  const float x1 = r[1] * 0.25f - 0.5f;
  const float y1 = r[2] * 0.25f - 0.5f;
  const float x2 = r[3] * 0.25f - 0.5f;
  const float y2 = r[4] * 0.25f - 0.5f;
  const float rw = fmaxf(x2 - x1, 0.1f);
  const float rh = fmaxf(y2 - y1, 0.1f);
  const float bw = rw * (1.0f / OUTD);
  const float bh = rh * (1.0f / OUTD);
  const float cy = y1 + (ph + 0.5f) * bh;
  const float cx = x1 + (pw + 0.5f) * bw;

  float sw = 1.f, sh = 1.f, ofw = 0.f, ofh = 0.f;
  if (off) {
    ofw = off[(size_t)n * offStride + 0 * SP + s];
    ofh = off[(size_t)n * offStride + 1 * SP + s];
  }
  if (tcv) {
    sw  = 1.f + 0.5f * tanhf(tcv[(size_t)n * tcStride + 0 * SP + s]);
    sh  = 1.f + 0.5f * tanhf(tcv[(size_t)n * tcStride + 1 * SP + s]);
    ofw += tcv[(size_t)n * tcStride + 2 * SP + s];
    ofh += tcv[(size_t)n * tcStride + 3 * SP + s];
  }
  const float ow = 0.1f * rw * ofw;          // GAMMA = 0.1
  const float oh = 0.1f * rh * ofh;

  const float* fb = feat_t + (size_t)b * HWSZ * CC;
  float acc = 0.f;
  #pragma unroll
  for (int iy = 0; iy < 2; ++iy) {
    const float gy = (iy + 0.5f) * 0.5f - 0.5f;       // {-0.25, 0.25}
    const float y  = cy + gy * bh * sh + oh;
    #pragma unroll
    for (int ix = 0; ix < 2; ++ix) {
      const float gx = (ix + 0.5f) * 0.5f - 0.5f;
      const float x  = cx + gx * bw * sw + ow;
      const bool valid = (y > -1.f) && (y < (float)HH) && (x > -1.f) && (x < (float)WWD);
      const float yc = fminf(fmaxf(y, 0.f), (float)(HH - 1));
      const float xc = fminf(fmaxf(x, 0.f), (float)(WWD - 1));
      const int y0 = (int)floorf(yc);
      const int x0 = (int)floorf(xc);
      const int y1i = min(y0 + 1, HH - 1);
      const int x1i = min(x0 + 1, WWD - 1);
      const float ly = yc - (float)y0, lx = xc - (float)x0;
      const float hy = 1.f - ly, hx = 1.f - lx;
      float v = hy * hx * fb[((size_t)y0  * WWD + x0 ) * CC + c]
              + hy * lx * fb[((size_t)y0  * WWD + x1i) * CC + c]
              + ly * hx * fb[((size_t)y1i * WWD + x0 ) * CC + c]
              + ly * lx * fb[((size_t)y1i * WWD + x1i) * CC + c];
      acc += valid ? v : 0.f;
    }
  }
  out[((size_t)n * SP + s) * CC + c] = acc * 0.25f;   // mean over 2x2 grid
}

// ---------------------------------------------------------------------------
// Weight pack: w[K, Nreal] f32 -> Bpack[N/16][K/32][32 lanes][16 bf16]
// B fragment: lane -> n = nt*16 + (lane&15); element e -> k = kt*32 + (e>>3)*16 + (lane>>4)*8 + (e&7)
// ---------------------------------------------------------------------------
__global__ __launch_bounds__(256) void pack_w(const float* __restrict__ w,
                                              __bf16* __restrict__ dst,
                                              int K, int Nreal, int Npad) {
  const size_t total = (size_t)(Npad >> 4) * (size_t)(K >> 5) * 512;
  size_t t = (size_t)blockIdx.x * 256 + threadIdx.x;
  if (t >= total) return;
  const int e    = (int)(t & 15);
  const int lane = (int)((t >> 4) & 31);
  const size_t t2 = t >> 9;
  const int KT = K >> 5;
  const int kt = (int)(t2 % KT);
  const int nt = (int)(t2 / KT);
  const int k = kt * 32 + ((e >> 3) << 4) + ((lane >> 4) << 3) + (e & 7);
  const int n = nt * 16 + (lane & 15);
  const float v = (n < Nreal) ? w[(size_t)k * Nreal + n] : 0.f;
  dst[t] = (__bf16)v;
}

// ---------------------------------------------------------------------------
// Activation pack from pooled [N,49,C] (flat k = c*49 + s), optional SE gate
// (x*(1+gate)). A fragment: m = mt*16+(lane&15); k = kt*32+(e>>3)*16+(lane>>4)*8+(e&7)
// ---------------------------------------------------------------------------
__global__ __launch_bounds__(256) void pack_act_pool(const float* __restrict__ pool,
                                                     const float* __restrict__ gate,
                                                     __bf16* __restrict__ dst) {
  size_t t = (size_t)blockIdx.x * 256 + threadIdx.x;       // total = 512*FLATK exactly
  const int e    = (int)(t & 15);
  const int lane = (int)((t >> 4) & 31);
  const size_t t2 = t >> 9;
  const int KT = FLATK >> 5;                                // 392
  const int kt = (int)(t2 % KT);
  const int mt = (int)(t2 / KT);
  const int m = mt * 16 + (lane & 15);
  const int k = kt * 32 + ((e >> 3) << 4) + ((lane >> 4) << 3) + (e & 7);
  const int c = k / SP;
  const int s = k - c * SP;
  float v = pool[((size_t)m * SP + s) * CC + c];
  if (gate) v *= (1.f + gate[(size_t)m * CC + c]);
  dst[t] = (__bf16)v;
}

// Activation pack from row-major [512, K] f32
__global__ __launch_bounds__(256) void pack_act_row(const float* __restrict__ src,
                                                    __bf16* __restrict__ dst, int K) {
  size_t t = (size_t)blockIdx.x * 256 + threadIdx.x;       // total = 512*K exactly
  const int e    = (int)(t & 15);
  const int lane = (int)((t >> 4) & 31);
  const size_t t2 = t >> 9;
  const int KT = K >> 5;
  const int kt = (int)(t2 % KT);
  const int mt = (int)(t2 / KT);
  const int m = mt * 16 + (lane & 15);
  const int k = kt * 32 + ((e >> 3) << 4) + ((lane >> 4) << 3) + (e & 7);
  dst[t] = (__bf16)src[(size_t)m * K + k];
}

// ---------------------------------------------------------------------------
// TDM helper: issue a 2D tensor_load_to_lds of `rows` rows x 512 bf16, with
// row stride `strideElems` (data_size units), destination LDS offset ldsOff.
// D# packing per CDNA5 ISA 8.3/8.4 (count=1, data_size=2B, type=2).
// This toolchain exposes the 6-arg builtin (extra SGPR group; zeros for 2D).
// ---------------------------------------------------------------------------
__device__ __forceinline__ void tdm_load_slab(const __bf16* gptr, unsigned ldsOff,
                                              unsigned rows, unsigned strideElems) {
  const unsigned long long ga = (unsigned long long)(uintptr_t)gptr;
  u32x4 g0;
  g0[0] = 1u;                                               // count=1, user mode
  g0[1] = ldsOff;                                           // lds_addr (bytes)
  g0[2] = (unsigned)ga;                                     // global_addr[31:0]
  g0[3] = (unsigned)((ga >> 32) & 0x1FFFFFFu) | (2u << 30); // global_addr[56:32] | type=2
  i32x8 g1;
  g1[0] = 0x00010000;                   // data_size=1 (2 bytes); no flags/pad
  g1[1] = (int)(512u << 16);            // tensor_dim0[15:0]=512
  g1[2] = (int)(rows << 16);            // tensor_dim0 hi=0 | tensor_dim1[15:0]=rows
  g1[3] = (int)(512u << 16);            // tensor_dim1 hi=0 | tile_dim0=512
  g1[4] = (int)rows;                    // tile_dim1=rows, tile_dim2=0
  g1[5] = (int)strideElems;             // tensor_dim0_stride[31:0] (elements)
  g1[6] = 0;                            // stride hi | tensor_dim1_stride lo
  g1[7] = 0;
  const i32x4 z4 = {0, 0, 0, 0};
  const i32x8 z8 = {0, 0, 0, 0, 0, 0, 0, 0};
  __builtin_amdgcn_tensor_load_to_lds(g0, g1, z4, z4, z8, 0);
}

// ---------------------------------------------------------------------------
// WMMA GEMM with TDM double-buffered LDS staging.
// C[512, Npad] = A[512,K] * B[K,Npad] (+bias, opt relu)
// A/B pre-swizzled to fragment layout [tile][kt][lane][16 bf16].
// 256 threads = 8 waves (2 along M x 4 along N); block tile 64x128; each wave a
// 32x32 C tile (2x2 WMMA frags). Per k-step, wave 0 DMAs the A slab (4x1KB) and
// B slab (8x1KB) into LDS via tensor_load_to_lds; waves read fragments with ds
// loads. Dynamic LDS = 24 KB (double buffer).
// grid = (512/64, Npad/128)
// ---------------------------------------------------------------------------
__global__ __launch_bounds__(256) void gemm_wmma_bf16(
    const __bf16* __restrict__ Apack, const __bf16* __restrict__ Bpack,
    const float* __restrict__ bias, float* __restrict__ Cout,
    int Npad, int K, int nbias, int relu) {
  extern __shared__ __align__(1024) char smem[];            // [A0 4K][A1 4K][B0 8K][B1 8K]
  const unsigned ldsBase = __builtin_amdgcn_groupstaticsize();  // dynamic-LDS base offset
  const unsigned offA[2] = {0u, 4096u};
  const unsigned offB[2] = {8192u, 16384u};

  const int lane = threadIdx.x & 31;
  const int wave = threadIdx.x >> 5;
  const int wm = wave & 1;          // 2 waves along M
  const int wn = wave >> 1;         // 4 waves along N
  const int m0 = blockIdx.x * 64 + wm * 32;
  const int n0 = blockIdx.y * 128 + wn * 32;
  const int KT = K >> 5;
  const size_t tstride = (size_t)KT * 512;                  // elements per tile-row

  const __bf16* aSlab = Apack + (size_t)(blockIdx.x * 4) * tstride;  // 4 m-tiles
  const __bf16* bSlab = Bpack + (size_t)(blockIdx.y * 8) * tstride;  // 8 n-tiles

  f32x8 acc00 = {}, acc01 = {}, acc10 = {}, acc11 = {};

  if (wave == 0) {                                          // prime buffer 0
    tdm_load_slab(aSlab, ldsBase + offA[0], 4u, (unsigned)tstride);
    tdm_load_slab(bSlab, ldsBase + offB[0], 8u, (unsigned)tstride);
  }

  for (int kt = 0; kt < KT; ++kt) {
    const int cur = kt & 1;
    const int nxt = cur ^ 1;
    if (wave == 0) __builtin_amdgcn_s_wait_tensorcnt(0);    // cur slabs resident
    __syncthreads();                                        // also: nxt free to overwrite
    if (wave == 0 && (kt + 1) < KT) {
      tdm_load_slab(aSlab + (size_t)(kt + 1) * 512, ldsBase + offA[nxt], 4u, (unsigned)tstride);
      tdm_load_slab(bSlab + (size_t)(kt + 1) * 512, ldsBase + offB[nxt], 8u, (unsigned)tstride);
    }
    const __bf16* sA = (const __bf16*)(smem + offA[cur]);
    const __bf16* sB = (const __bf16*)(smem + offB[cur]);
    bf16x16 a0 = *(const bf16x16*)(sA + (size_t)(wm * 2 + 0) * 512 + lane * 16);
    bf16x16 a1 = *(const bf16x16*)(sA + (size_t)(wm * 2 + 1) * 512 + lane * 16);
    bf16x16 b0 = *(const bf16x16*)(sB + (size_t)(wn * 2 + 0) * 512 + lane * 16);
    bf16x16 b1 = *(const bf16x16*)(sB + (size_t)(wn * 2 + 1) * 512 + lane * 16);
    acc00 = __builtin_amdgcn_wmma_f32_16x16x32_bf16(false, a0, false, b0, (short)0, acc00, false, false);
    acc01 = __builtin_amdgcn_wmma_f32_16x16x32_bf16(false, a0, false, b1, (short)0, acc01, false, false);
    acc10 = __builtin_amdgcn_wmma_f32_16x16x32_bf16(false, a1, false, b0, (short)0, acc10, false, false);
    acc11 = __builtin_amdgcn_wmma_f32_16x16x32_bf16(false, a1, false, b1, (short)0, acc11, false, false);
  }

  const int ln = lane & 15;
  const int lh = lane >> 4;
  f32x8 accs[2][2] = {{acc00, acc01}, {acc10, acc11}};
  #pragma unroll
  for (int i = 0; i < 2; ++i)
    #pragma unroll
    for (int j = 0; j < 2; ++j) {
      const int n = n0 + j * 16 + ln;
      const float bv = (n < nbias) ? bias[n] : 0.f;
      const int mb = m0 + i * 16 + lh * 8;
      #pragma unroll
      for (int rr = 0; rr < 8; ++rr) {
        float v = accs[i][j][rr] + bv;
        if (relu) v = fmaxf(v, 0.f);
        Cout[(size_t)(mb + rr) * Npad + n] = v;
      }
    }
}

// ---------------------------------------------------------------------------
// SE: spatial mean of [N,49,C] -> [N,C]
// ---------------------------------------------------------------------------
__global__ __launch_bounds__(256) void se_mean(const float* __restrict__ pool,
                                               float* __restrict__ ymean) {
  const int n = blockIdx.x, c = threadIdx.x;
  float s = 0.f;
  for (int t = 0; t < SP; ++t) s += pool[((size_t)n * SP + t) * CC + c];
  ymean[(size_t)n * CC + c] = s * (1.f / (float)SP);
}

// SE gate: sigmoid(relu(y@w1)@w2). w1 [C,16], w2 [16,C]. One block per roi.
__global__ __launch_bounds__(256) void se_gate(const float* __restrict__ ymean,
                                               const float* __restrict__ w1,
                                               const float* __restrict__ w2,
                                               float* __restrict__ gate) {
  __shared__ float ys[CC];
  __shared__ float hs[16];
  const int n = blockIdx.x, tid = threadIdx.x;
  ys[tid] = ymean[(size_t)n * CC + tid];
  __syncthreads();
  if (tid < 16) {
    float a = 0.f;
    for (int c = 0; c < CC; ++c) a += ys[c] * w1[c * 16 + tid];
    hs[tid] = fmaxf(a, 0.f);
  }
  __syncthreads();
  float a = 0.f;
  #pragma unroll
  for (int j = 0; j < 16; ++j) a += hs[j] * w2[j * CC + tid];
  gate[(size_t)n * CC + tid] = 1.f / (1.f + expf(-a));
}

// pooled [N,49,C] -> d_out [N,C,7,7] (coalesced writes)
__global__ __launch_bounds__(256) void pool_to_nchw(const float* __restrict__ pool,
                                                    float* __restrict__ out) {
  size_t t = (size_t)blockIdx.x * 256 + threadIdx.x;       // total = 512*256*49
  const size_t rem = t % ((size_t)CC * SP);
  const int n = (int)(t / ((size_t)CC * SP));
  const int c = (int)(rem / SP);
  const int s = (int)(rem % SP);
  out[t] = pool[((size_t)n * SP + s) * CC + c];
}

// ---------------------------------------------------------------------------
// host-side MLP driver (all launches on `stream`)
// ---------------------------------------------------------------------------
#define GEMM_LDS 24576

static void run_mlp(const float* w1, const float* b1, const float* w2, const float* b2,
                    const float* w3, const float* b3, int nout, int npad3,
                    const float* poolsrc, const float* gate,
                    __bf16* ApackBig, __bf16* ApackH,
                    __bf16* Wp1, __bf16* Wp2, __bf16* Wp3,
                    float* h1, float* h2, float* outbuf, hipStream_t stream) {
  // layer 1: [512,FLATK] @ [FLATK,1024]
  pack_w<<<(int)(((size_t)(FC >> 4) * (FLATK >> 5) * 512) / 256), 256, 0, stream>>>(w1, Wp1, FLATK, FC, FC);
  pack_act_pool<<<(int)(((size_t)ROIS_N * FLATK) / 256), 256, 0, stream>>>(poolsrc, gate, ApackBig);
  gemm_wmma_bf16<<<dim3(ROIS_N / 64, FC / 128), 256, GEMM_LDS, stream>>>(ApackBig, Wp1, b1, h1, FC, FLATK, FC, 1);
  // layer 2: [512,1024] @ [1024,1024]
  pack_w<<<(int)(((size_t)(FC >> 4) * (FC >> 5) * 512) / 256), 256, 0, stream>>>(w2, Wp2, FC, FC, FC);
  pack_act_row<<<(int)(((size_t)ROIS_N * FC) / 256), 256, 0, stream>>>(h1, ApackH, FC);
  gemm_wmma_bf16<<<dim3(ROIS_N / 64, FC / 128), 256, GEMM_LDS, stream>>>(ApackH, Wp2, b2, h2, FC, FC, FC, 1);
  // layer 3: [512,1024] @ [1024,nout(pad)]
  pack_w<<<(int)(((size_t)(npad3 >> 4) * (FC >> 5) * 512) / 256), 256, 0, stream>>>(w3, Wp3, FC, nout, npad3);
  pack_act_row<<<(int)(((size_t)ROIS_N * FC) / 256), 256, 0, stream>>>(h2, ApackH, FC);
  gemm_wmma_bf16<<<dim3(ROIS_N / 64, npad3 / 128), 256, GEMM_LDS, stream>>>(ApackH, Wp3, b3, outbuf, npad3, FC, nout, 0);
}

extern "C" void kernel_launch(void* const* d_in, const int* in_sizes, int n_in,
                              void* d_out, int out_size, void* d_ws, size_t ws_size,
                              hipStream_t stream) {
  (void)in_sizes; (void)n_in; (void)out_size; (void)ws_size;
  // d_in order = setup_inputs dict flattening:
  // 0 feat, 1 rois,
  // params: off{w1,b1,w2,b2,w3,b3}=2..7, cls{..}=8..13, reg{..}=14..19,
  //         se0_w1=20, se0_w2=21, se1_w1=22, se1_w2=23
  const float* feat = (const float*)d_in[0];
  const float* rois = (const float*)d_in[1];
  const float* off_w1 = (const float*)d_in[2];
  const float* off_b1 = (const float*)d_in[3];
  const float* off_w2 = (const float*)d_in[4];
  const float* off_b2 = (const float*)d_in[5];
  const float* off_w3 = (const float*)d_in[6];
  const float* off_b3 = (const float*)d_in[7];
  const float* cls_w1 = (const float*)d_in[8];
  const float* cls_b1 = (const float*)d_in[9];
  const float* cls_w2 = (const float*)d_in[10];
  const float* cls_b2 = (const float*)d_in[11];
  const float* cls_w3 = (const float*)d_in[12];
  const float* cls_b3 = (const float*)d_in[13];
  const float* reg_w1 = (const float*)d_in[14];
  const float* reg_b1 = (const float*)d_in[15];
  const float* reg_w2 = (const float*)d_in[16];
  const float* reg_b2 = (const float*)d_in[17];
  const float* reg_w3 = (const float*)d_in[18];
  const float* reg_b3 = (const float*)d_in[19];
  const float* se0_w1 = (const float*)d_in[20];
  const float* se0_w2 = (const float*)d_in[21];
  const float* se1_w1 = (const float*)d_in[22];
  const float* se1_w2 = (const float*)d_in[23];

  // workspace bump allocator (~183 MB total)
  char* ws = (char*)d_ws;
  auto alloc = [&](size_t bytes) -> void* {
    void* p = (void*)ws;
    ws += (bytes + 255) & ~(size_t)255;
    return p;
  };
  float*  feat_t   = (float*) alloc((size_t)BB * HWSZ * CC * 4);          // 81.9 MB
  float*  pool0    = (float*) alloc((size_t)ROIS_N * SP * CC * 4);        // 25.7 MB
  float*  pool1    = (float*) alloc((size_t)ROIS_N * SP * CC * 4);        // 25.7 MB
  __bf16* ApackBig = (__bf16*)alloc((size_t)ROIS_N * FLATK * 2);          // 12.85 MB
  __bf16* ApackH   = (__bf16*)alloc((size_t)ROIS_N * FC * 2);             // 1 MB
  __bf16* Wp1      = (__bf16*)alloc((size_t)FLATK * FC * 2);              // 25.7 MB
  __bf16* Wp2      = (__bf16*)alloc((size_t)FC * FC * 2);                 // 2 MB
  __bf16* Wp3      = (__bf16*)alloc((size_t)FC * 256 * 2);                // 0.5 MB
  float*  h1       = (float*) alloc((size_t)ROIS_N * FC * 4);
  float*  h2       = (float*) alloc((size_t)ROIS_N * FC * 4);
  float*  off_out  = (float*) alloc((size_t)ROIS_N * 128 * 4);
  float*  tc_out   = (float*) alloc((size_t)ROIS_N * 256 * 4);
  float*  tr_out   = (float*) alloc((size_t)ROIS_N * 256 * 4);
  float*  ymean    = (float*) alloc((size_t)ROIS_N * CC * 4);
  float*  gate0    = (float*) alloc((size_t)ROIS_N * CC * 4);
  float*  gate1    = (float*) alloc((size_t)ROIS_N * CC * 4);

  const int poolBlocks = ROIS_N * SP;                              // 25088
  const int nscBlocks  = (int)(((size_t)ROIS_N * CC * SP) / 256);  // 25088

  // 1. feat -> NHWC
  feat_transpose<<<dim3(HWSZ / 32, CC / 32, BB), 256, 0, stream>>>(feat, feat_t);

  // 2. plain pool -> x
  deform_pool<<<poolBlocks, 256, 0, stream>>>(feat_t, rois, nullptr, 0, nullptr, 0, pool0);

  // 3. offset MLP (nout=98, pad to 128)
  run_mlp(off_w1, off_b1, off_w2, off_b2, off_w3, off_b3, 98, 128,
          pool0, nullptr, ApackBig, ApackH, Wp1, Wp2, Wp3, h1, h2, off_out, stream);

  // 4. pool with offsets -> xd
  deform_pool<<<poolBlocks, 256, 0, stream>>>(feat_t, rois, off_out, 128, nullptr, 0, pool1);

  // 5. SE gates
  se_mean<<<ROIS_N, 256, 0, stream>>>(pool1, ymean);
  se_gate<<<ROIS_N, 256, 0, stream>>>(ymean, se0_w1, se0_w2, gate0);
  se_gate<<<ROIS_N, 256, 0, stream>>>(ymean, se1_w1, se1_w2, gate1);

  // 6. cls / reg MLPs (gated inputs), nout=196 pad 256
  run_mlp(cls_w1, cls_b1, cls_w2, cls_b2, cls_w3, cls_b3, 196, 256,
          pool1, gate0, ApackBig, ApackH, Wp1, Wp2, Wp3, h1, h2, tc_out, stream);
  run_mlp(reg_w1, reg_b1, reg_w2, reg_b2, reg_w3, reg_b3, 196, 256,
          pool1, gate1, ApackBig, ApackH, Wp1, Wp2, Wp3, h1, h2, tr_out, stream);

  // 7. final deformable pools -> outputs
  float* out_f = (float*)d_out;
  deform_pool<<<poolBlocks, 256, 0, stream>>>(feat_t, rois, off_out, 128, tc_out, 256, pool0);
  pool_to_nchw<<<nscBlocks, 256, 0, stream>>>(pool0, out_f);
  deform_pool<<<poolBlocks, 256, 0, stream>>>(feat_t, rois, off_out, 128, tr_out, 256, pool0);
  pool_to_nchw<<<nscBlocks, 256, 0, stream>>>(pool0, out_f + (size_t)ROIS_N * CC * SP);
}
